// PointNetSetAbstraction_46557445488933
// MI455X (gfx1250) — matrix operand order repaired and checked by reading, not compiled, not run on hardware
//
#include <hip/hip_runtime.h>
#include <hip/hip_bf16.h>

// ---------------- CDNA5 WMMA types ----------------
typedef __attribute__((ext_vector_type(16))) __bf16          v16bf;
typedef __attribute__((ext_vector_type(8)))  float           v8f;

union Frag { uint4 q[2]; v16bf b; };     // 32 bytes = one 16x32-bf16 wmma operand per lane

#define NPTS    16384
#define BATCH   8
#define NPOINT  512
#define NSAMPLE 32
#define RADIUS2 0.16f
#define MROWS   (BATCH*NPOINT*NSAMPLE)   // 131072
#define SG      96                        // grouped stride (67 zero-padded to 3x32)

// RNE float -> bf16 bits
__device__ __forceinline__ unsigned short f2bf(float f) {
    unsigned int u = __float_as_uint(f);
    unsigned int r = u + 0x7FFFu + ((u >> 16) & 1u);
    return (unsigned short)(r >> 16);
}

// ---------------- 1) Farthest point sampling: one block per batch ----------------
__global__ void fps_kernel(const float* __restrict__ xyz, float* __restrict__ dists,
                           float* __restrict__ fpsXyz, float* __restrict__ outFps)
{
    const int b   = blockIdx.x;
    const int tid = threadIdx.x;                     // 256 threads
    __shared__ float rv[256];
    __shared__ int   ri[256];
    __shared__ float cur[3];
    __shared__ int   s_next;
    const float* px = xyz + (size_t)b * NPTS * 3;
    float* dd = dists + (size_t)b * NPTS;
    for (int j = tid; j < NPTS; j += 256) dd[j] = 1e10f;
    if (tid < 3) {
        cur[tid] = px[tid];
        fpsXyz[((size_t)b * NPOINT) * 3 + tid] = px[tid];
        outFps[((size_t)b * NPOINT) * 3 + tid] = px[tid];
    }
    __syncthreads();
    for (int s = 1; s < NPOINT; ++s) {
        const float cx = cur[0], cy = cur[1], cz = cur[2];
        float best = -1.0f; int bi = 0;
        for (int j = tid; j < NPTS; j += 256) {
            float dx = px[j*3+0] - cx, dy = px[j*3+1] - cy, dz = px[j*3+2] - cz;
            float d = dx*dx + dy*dy + dz*dz;
            float od = dd[j];
            d = d < od ? d : od;
            dd[j] = d;
            if (d > best) { best = d; bi = j; }
        }
        rv[tid] = best; ri[tid] = bi;
        __syncthreads();
        for (int t = 128; t > 0; t >>= 1) {
            if (tid < t && rv[tid + t] > rv[tid]) { rv[tid] = rv[tid + t]; ri[tid] = ri[tid + t]; }
            __syncthreads();
        }
        if (tid == 0) s_next = ri[0];
        __syncthreads();
        const int ni = s_next;
        if (tid < 3) {
            float v = px[ni*3 + tid];
            cur[tid] = v;
            fpsXyz[((size_t)b * NPOINT + s) * 3 + tid] = v;
            outFps[((size_t)b * NPOINT + s) * 3 + tid] = v;
        }
        __syncthreads();
    }
}

// ---------------- 2) kNN (sorted 32-NN) + radius fill + grouping. One block/query ----------------
// Dynamic LDS: d2[16384] + rv[256] + ri[256] + chosen[32] + chosend[32]  (~68KB, fine on 320KB/WGP)
__global__ void knn_group_kernel(const float* __restrict__ xyz, const float* __restrict__ feats,
                                 const float* __restrict__ fpsXyz, float* __restrict__ grouped)
{
    extern __shared__ __align__(16) char smem_raw[];
    float* d2      = (float*)smem_raw;               // 16384
    float* rv      = d2 + NPTS;                      // 256
    int*   ri      = (int*)(rv + 256);               // 256
    int*   chosen  = ri + 256;                       // 32
    float* chosend = (float*)(chosen + 32);          // 32

    const int q   = blockIdx.x;                      // 0..4095
    const int b   = q >> 9;
    const int tid = threadIdx.x;                     // 256 threads
    const float cx = fpsXyz[q*3+0], cy = fpsXyz[q*3+1], cz = fpsXyz[q*3+2];
    const float* px = xyz   + (size_t)b * NPTS * 3;
    const float* pf = feats + (size_t)b * NPTS * 64;

    for (int j = tid; j < NPTS; j += 256) {
        float dx = px[j*3+0] - cx, dy = px[j*3+1] - cy, dz = px[j*3+2] - cz;
        d2[j] = dx*dx + dy*dy + dz*dz;
    }
    __syncthreads();
    // 32 rounds of block argmin -> sorted nearest indices (matches top_k order)
    for (int r = 0; r < NSAMPLE; ++r) {
        float best = 3.4e38f; int bi = 0;
        for (int j = tid; j < NPTS; j += 256) {
            float v = d2[j];
            if (v < best) { best = v; bi = j; }
        }
        rv[tid] = best; ri[tid] = bi;
        __syncthreads();
        for (int t = 128; t > 0; t >>= 1) {
            if (tid < t && rv[tid + t] < rv[tid]) { rv[tid] = rv[tid + t]; ri[tid] = ri[tid + t]; }
            __syncthreads();
        }
        if (tid == 0) {
            int w = ri[0];
            chosen[r] = w; chosend[r] = rv[0];
            d2[w] = 3.4e38f;                          // exclude from later rounds
        }
        __syncthreads();
    }
    if (tid < NSAMPLE && chosend[tid] > RADIUS2) chosen[tid] = chosen[0];  // radius fill w/ nearest
    __syncthreads();
    // grouped row: [dx,dy,dz, 64 feats, zero pad to 96]
    for (int t = tid; t < NSAMPLE * SG; t += 256) {
        int r = t / SG, c = t - r * SG;
        int idx = chosen[r];
        float v;
        if (c < 3)       v = px[idx*3 + c] - ((c == 0) ? cx : (c == 1) ? cy : cz);
        else if (c < 67) v = pf[(size_t)idx * 64 + (c - 3)];
        else             v = 0.0f;
        grouped[((size_t)q * NSAMPLE + r) * SG + c] = v;
    }
}

// ---------------- 3) GEMM (bf16 WMMA) + bias + channel stats; optional fused input BN+ReLU ----------------
// Compile-time shapes: fully unrolled k-loop, LDS-resident zero-padded W, 16B-aligned fragment loads.
// Block = 128 threads (4 waves). Wave w -> 16x16 tile (Mtile=blockIdx.x, Ntile=blockIdx.y*4+w).
// X stride == KD (96 for layer0 zero-padded grouped, 64 otherwise).
template<int CIN, int KD, int COUT, bool HAS_BN>
__global__ void gemm_bn_kernel(const float* __restrict__ X,
                               const float* __restrict__ inScale, const float* __restrict__ inShift,
                               const float* __restrict__ W, const float* __restrict__ bias,
                               float* __restrict__ Y,
                               float* __restrict__ chSum, float* __restrict__ chSq)
{
    constexpr int KSTEPS = KD / 32;
    __shared__ unsigned short aTile[16 * 32];        // bf16 A tile
    __shared__ unsigned short wTile[COUT * KD];      // bf16 W, zero-padded K

    const int tid   = threadIdx.x;                   // 0..127
    const int wave  = tid >> 5;
    const int lane  = tid & 31;
    const int half  = lane >> 4;
    const int mBase = blockIdx.x * 16;
    const int nCol  = (blockIdx.y * 4 + wave) * 16 + (lane & 15);
    const int mRow  = lane & 15;

    // stage W once (bf16, K zero-padded to KD); COUT*KD divisible by 128
    for (int t = tid; t < COUT * KD; t += 128) {
        int o = t / KD, k = t - o * KD;
        float wv = (k < CIN) ? W[o * CIN + k] : 0.0f;
        wTile[t] = f2bf(wv);
    }

    v8f acc = {};
    #pragma unroll
    for (int ks = 0; ks < KSTEPS; ++ks) {
        __syncthreads();                             // (iter 0: also covers wTile staging)
        // stage A 16x32: one float4 per thread, fused BN+ReLU of previous layer
        {
            const int m = tid >> 3;
            const int k = (tid & 7) * 4;
            float4 v = *(const float4*)&X[(size_t)(mBase + m) * KD + ks * 32 + k];
            if (HAS_BN) {
                const int kg = ks * 32 + k;
                v.x = fmaxf(v.x * inScale[kg + 0] + inShift[kg + 0], 0.0f);
                v.y = fmaxf(v.y * inScale[kg + 1] + inShift[kg + 1], 0.0f);
                v.z = fmaxf(v.z * inScale[kg + 2] + inShift[kg + 2], 0.0f);
                v.w = fmaxf(v.w * inScale[kg + 3] + inShift[kg + 3], 0.0f);
            }
            unsigned long long pk =
                  (unsigned long long)f2bf(v.x)
                | ((unsigned long long)f2bf(v.y) << 16)
                | ((unsigned long long)f2bf(v.z) << 32)
                | ((unsigned long long)f2bf(v.w) << 48);
            *(unsigned long long*)&aTile[m * 32 + k] = pk;   // 8B-aligned ds_store_b64
        }
        __syncthreads();
        // A fragment: lanes 0-15 K=0..15; lanes 16-31 K=16..31 (matches B order)
        Frag a, b;
        a.q[0] = *(const uint4*)&aTile[mRow * 32 + half * 16];        // 16B-aligned ds_load_b128
        a.q[1] = *(const uint4*)&aTile[mRow * 32 + half * 16 + 8];
        // B fragment: col = nCol; lanes 0-15 K=0..15, lanes 16-31 K=16..31 of this kstep
        b.q[0] = *(const uint4*)&wTile[nCol * KD + ks * 32 + half * 16];
        b.q[1] = *(const uint4*)&wTile[nCol * KD + ks * 32 + half * 16 + 8];
        acc = __builtin_amdgcn_wmma_f32_16x16x32_bf16(false, a.b, false, b.b, (short)0, acc, false, false);
    }

    // bias, store pre-BN activation, accumulate per-channel stats
    const float bv = bias[nCol];
    float s = 0.0f, sq = 0.0f;
    #pragma unroll
    for (int r = 0; r < 8; ++r) {
        int m = r + half * 8;                        // C/D layout: VGPR r -> M=r (lo lanes), M=8+r (hi lanes)
        float y = acc[r] + bv;
        Y[(size_t)(mBase + m) * COUT + nCol] = y;
        s += y; sq += y * y;
    }
    s  += __shfl_xor(s, 16, 32);                     // combine the two lanes holding same column
    sq += __shfl_xor(sq, 16, 32);
    if (half == 0) { atomicAdd(&chSum[nCol], s); atomicAdd(&chSq[nCol], sq); }
}

// ---------------- 4) BN finalize: per-channel scale/shift ----------------
__global__ void bn_finalize_kernel(const float* __restrict__ sum, const float* __restrict__ sq,
                                   const float* __restrict__ g, const float* __restrict__ beta,
                                   float* __restrict__ scale, float* __restrict__ shift,
                                   int C, float invM)
{
    int o = threadIdx.x;
    if (o < C) {
        float mean = sum[o] * invM;
        float var  = sq[o] * invM - mean * mean;
        float sc   = g[o] * rsqrtf(var + 1e-5f);
        scale[o] = sc;
        shift[o] = beta[o] - mean * sc;
    }
}

// ---------------- 5) fused BN+ReLU + max over K=32 ----------------
__global__ void maxpool_kernel(const float* __restrict__ y2, const float* __restrict__ scale,
                               const float* __restrict__ shift, float* __restrict__ out)
{
    const int q = blockIdx.x;                        // 0..4095
    const int o = threadIdx.x;                       // 128 channels
    const float sc = scale[o], sh = shift[o];
    float acc = 0.0f;                                // relu output >= 0
    #pragma unroll 4
    for (int k = 0; k < NSAMPLE; ++k) {
        float v = y2[((size_t)q * NSAMPLE + k) * 128 + o] * sc + sh;
        v = v > 0.0f ? v : 0.0f;
        acc = acc > v ? acc : v;
    }
    out[(size_t)BATCH * NPOINT * 3 + (size_t)q * 128 + o] = acc;
}

// ---------------- workspace layout (floats) ----------------
#define OFF_DIST    0                                // 131072
#define OFF_SUM     131072                           // 768  (sum,sq for 3 layers x 128)
#define OFF_BN      131840                           // 768  (scale,shift x3 x 128)
#define OFF_FPS     132608                           // 12288
#define OFF_GROUPED 144896                           // 131072*96 = 12582912
#define OFF_Y0      12727808                         // 131072*64 = 8388608
#define OFF_Y1      21116416                         // 8388608
#define OFF_Y2      OFF_GROUPED                      // 131072*128 = 16777216 (reuses dead grouped+y0)

extern "C" void kernel_launch(void* const* d_in, const int* in_sizes, int n_in,
                              void* d_out, int out_size, void* d_ws, size_t ws_size,
                              hipStream_t stream) {
    const float* xyz   = (const float*)d_in[0];
    const float* feats = (const float*)d_in[1];
    const float* W0 = (const float*)d_in[2];  const float* b0 = (const float*)d_in[3];
    const float* g0 = (const float*)d_in[4];  const float* be0 = (const float*)d_in[5];
    const float* W1 = (const float*)d_in[6];  const float* b1 = (const float*)d_in[7];
    const float* g1 = (const float*)d_in[8];  const float* be1 = (const float*)d_in[9];
    const float* W2 = (const float*)d_in[10]; const float* b2 = (const float*)d_in[11];
    const float* g2 = (const float*)d_in[12]; const float* be2 = (const float*)d_in[13];
    float* out = (float*)d_out;
    float* ws  = (float*)d_ws;

    float* dists   = ws + OFF_DIST;
    float* sum0 = ws + OFF_SUM;       float* sq0 = sum0 + 128;
    float* sum1 = sq0 + 128;          float* sq1 = sum1 + 128;
    float* sum2 = sq1 + 128;          float* sq2 = sum2 + 128;
    float* sc0  = ws + OFF_BN;        float* sh0 = sc0 + 128;
    float* sc1  = sh0 + 128;          float* sh1 = sc1 + 128;
    float* sc2  = sh1 + 128;          float* sh2 = sc2 + 128;
    float* fpsXyz  = ws + OFF_FPS;
    float* grouped = ws + OFF_GROUPED;
    float* y0      = ws + OFF_Y0;
    float* y1      = ws + OFF_Y1;
    float* y2      = ws + OFF_Y2;

    hipMemsetAsync(sum0, 0, 768 * sizeof(float), stream);

    // 1) FPS
    fps_kernel<<<BATCH, 256, 0, stream>>>(xyz, dists, fpsXyz, out);

    // 2) kNN + group
    const size_t shmem = (size_t)(NPTS + 256) * sizeof(float) + (256 + 32) * sizeof(int) + 32 * sizeof(float);
    knn_group_kernel<<<BATCH * NPOINT, 256, shmem, stream>>>(xyz, feats, fpsXyz, grouped);

    const float invM = 1.0f / (float)MROWS;
    const int mtiles = MROWS / 16;                   // 8192

    // 3) layer 0: [M,67] x [67->64], K zero-padded to 96 (3 unrolled WMMA ksteps)
    gemm_bn_kernel<67, 96, 64, false><<<dim3(mtiles, 1), 128, 0, stream>>>(
        grouped, nullptr, nullptr, W0, b0, y0, sum0, sq0);
    bn_finalize_kernel<<<1, 64, 0, stream>>>(sum0, sq0, g0, be0, sc0, sh0, 64, invM);

    // 4) layer 1: fused BN0+ReLU on load
    gemm_bn_kernel<64, 64, 64, true><<<dim3(mtiles, 1), 128, 0, stream>>>(
        y0, sc0, sh0, W1, b1, y1, sum1, sq1);
    bn_finalize_kernel<<<1, 64, 0, stream>>>(sum1, sq1, g1, be1, sc1, sh1, 64, invM);

    // 5) layer 2: 128 outputs
    gemm_bn_kernel<64, 64, 128, true><<<dim3(mtiles, 2), 128, 0, stream>>>(
        y1, sc1, sh1, W2, b2, y2, sum2, sq2);
    bn_finalize_kernel<<<1, 128, 0, stream>>>(sum2, sq2, g2, be2, sc2, sh2, 128, invM);

    // 6) fused BN2+ReLU + max over K
    maxpool_kernel<<<BATCH * NPOINT, 128, 0, stream>>>(y2, sc2, sh2, out);
}
